// TorchRPTAttention_76295799046200
// MI455X (gfx1250) — matrix operand order repaired and checked by reading, not compiled, not run on hardware
//
#include <hip/hip_runtime.h>
#include <hip/hip_bf16.h>

// MI455X / gfx1250: bf16 WMMA pipeline for fused attention block.
// X,W -> bf16; Q/K/V = WMMA GEMMs (128x128 block, 32x64 wave tiles);
// RoPE(+scale) fused elementwise; flash attention with TDM-staged K,
// WMMA QK^T and PV; final WMMA GEMM -> f32 d_out.

typedef __bf16 bf16;
typedef __bf16 v16bf __attribute__((ext_vector_type(16)));
typedef float  v8f   __attribute__((ext_vector_type(8)));
typedef unsigned int v4u __attribute__((ext_vector_type(4)));
typedef int v8i __attribute__((ext_vector_type(8)));
typedef int v4i __attribute__((ext_vector_type(4)));

#define S_LEN  2048
#define HDIM   4096
#define NHEADS 32
#define NKVH   8
#define HEADD  128

// Tensor Data Mover availability (this toolchain: 6-arg builtin)
#if defined(__has_builtin)
#if __has_builtin(__builtin_amdgcn_tensor_load_to_lds) && \
    __has_builtin(__builtin_amdgcn_s_wait_tensorcnt)
#define USE_TDM 1
#else
#define USE_TDM 0
#endif
#else
#define USE_TDM 0
#endif

union FragU { v16bf v; uint4 u[2]; };
union H8    { uint4 u; bf16 h[8]; };

__device__ __forceinline__ v16bf load_frag2(const bf16* p0, const bf16* p1) {
  FragU f;
  f.u[0] = *reinterpret_cast<const uint4*>(p0);
  f.u[1] = *reinterpret_cast<const uint4*>(p1);
  return f.v;
}

__device__ __forceinline__ v8f wmma_bf16(v16bf a, v16bf b, v8f c) {
  // D = A(16x32 bf16) x B(32x16 bf16) + C(16x16 f32)
  return __builtin_amdgcn_wmma_f32_16x16x32_bf16(false, a, false, b,
                                                 (short)0, c, false, false);
}

// ---------------------------------------------------------------- convert
__global__ void k_f32_to_bf16(const float* __restrict__ in,
                              bf16* __restrict__ out, int n4) {
  int i = blockIdx.x * blockDim.x + threadIdx.x;
  if (i >= n4) return;
  float4 f = reinterpret_cast<const float4*>(in)[i];
  union { uint2 u; bf16 h[4]; } o;
  o.h[0] = (bf16)f.x; o.h[1] = (bf16)f.y;
  o.h[2] = (bf16)f.z; o.h[3] = (bf16)f.w;
  reinterpret_cast<uint2*>(out)[i] = o.u;
}

// ---------------------------------------------------------------- RoPE (+scale)
__global__ void k_rope(bf16* __restrict__ x, const float* __restrict__ cosb,
                       const float* __restrict__ sinb, int nheads, float scale) {
  int idx = blockIdx.x * blockDim.x + threadIdx.x;
  int i = idx & 63;             // freq index (HEADD/2 = 64)
  int t = idx >> 6;
  int h = t % nheads;
  int s = t / nheads;
  if (s >= S_LEN) return;
  bf16* p = x + ((size_t)s * nheads + h) * HEADD + 2 * i;
  float xr = (float)p[0], xi = (float)p[1];
  float c  = cosb[s * 64 + i], sn = sinb[s * 64 + i];
  p[0] = (bf16)((xr * c - xi * sn) * scale);
  p[1] = (bf16)((xr * sn + xi * c) * scale);
}

// ---------------------------------------------------------------- GEMM
// C[M x N] = A[M x K] * B[K x N], bf16 in, f32 accum. 256 thr = 8 waves.
// Block tile 128x128, wave tile 32x64 (2x4 WMMA 16x16x32 frags, 8 WMMA/step).
template <bool OUT_F32>
__global__ __launch_bounds__(256) void k_gemm_bf16(
    const bf16* __restrict__ A, const bf16* __restrict__ B,
    void* __restrict__ Cout, int M, int N, int K) {
  __shared__ bf16 As[128 * 40];   // row-major, padded stride 40
  __shared__ bf16 Bt[128 * 40];   // B tile transposed: [n][k]

  const int tid  = threadIdx.x;
  const int m0   = blockIdx.y * 128;
  const int n0   = blockIdx.x * 128;
  const int wave = tid >> 5, lane = tid & 31;
  const int half = lane >> 4, lrow = lane & 15;
  const int wm   = wave >> 1, wn = wave & 1;

  const int arow = tid >> 1, aseg = tid & 1;   // A staging: 128 rows x 2 segs
  const int bk   = tid >> 3, bn   = tid & 7;   // B staging: 32 k x 8 n-segs(16)

  v8f acc[2][4] = {};

  for (int k0 = 0; k0 < K; k0 += 32) {
    // ---- stage A tile (128x32)
    {
      const bf16* ga = A + (size_t)(m0 + arow) * K + k0 + aseg * 16;
      uint4 a0 = reinterpret_cast<const uint4*>(ga)[0];
      uint4 a1 = reinterpret_cast<const uint4*>(ga)[1];
      uint4* dst = reinterpret_cast<uint4*>(&As[arow * 40 + aseg * 16]);
      dst[0] = a0; dst[1] = a1;
      if (k0 + 32 < K)
        __builtin_prefetch(ga + 32, 0, 1);     // global_prefetch_b8
    }
    // ---- stage B tile transposed (32x128 -> Bt[128][32])
    {
      const bf16* gb = B + (size_t)(k0 + bk) * N + n0 + bn * 16;
      H8 b0, b1;
      b0.u = reinterpret_cast<const uint4*>(gb)[0];
      b1.u = reinterpret_cast<const uint4*>(gb)[1];
#pragma unroll
      for (int i = 0; i < 8; ++i) {
        Bt[(bn * 16 + i) * 40 + bk]     = b0.h[i];
        Bt[(bn * 16 + 8 + i) * 40 + bk] = b1.h[i];
      }
    }
    __syncthreads();

    v16bf af[2], bfm[4];
#pragma unroll
    for (int t = 0; t < 2; ++t) {
      const bf16* pa = &As[(wm * 32 + t * 16 + lrow) * 40 + half * 8];
      af[t] = load_frag2(pa, pa + 16);
    }
#pragma unroll
    for (int t = 0; t < 4; ++t) {
      const bf16* pb = &Bt[(wn * 64 + t * 16 + lrow) * 40 + half * 8];
      bfm[t] = load_frag2(pb, pb + 16);
    }
#pragma unroll
    for (int ti = 0; ti < 2; ++ti)
#pragma unroll
      for (int tj = 0; tj < 4; ++tj)
        acc[ti][tj] = wmma_bf16(af[ti], bfm[tj], acc[ti][tj]);
    __syncthreads();
  }

  // ---- epilogue (C layout: m = v + 8*half, n = lane%16)
#pragma unroll
  for (int ti = 0; ti < 2; ++ti)
#pragma unroll
    for (int tj = 0; tj < 4; ++tj)
#pragma unroll
      for (int v = 0; v < 8; ++v) {
        int row = m0 + wm * 32 + ti * 16 + v + 8 * half;
        int col = n0 + wn * 64 + tj * 16 + lrow;
        float val = acc[ti][tj][v];
        if (OUT_F32)
          reinterpret_cast<float*>(Cout)[(size_t)row * N + col] = val;
        else
          reinterpret_cast<bf16*>(Cout)[(size_t)row * N + col] = (bf16)val;
      }
}

// ---------------------------------------------------------------- flash attention
// grid = (S/64, NHEADS), 128 threads (4 waves). Wave owns 16 q rows.
// Key blocks of 32; causal skip; online softmax; WMMA QK^T and PV.
// K block staged into LDS by the Tensor Data Mover (padded rows: 272B stride).
#define KROW 136   // Kst row stride in bf16 elements (256B data + 16B pad)

__global__ __launch_bounds__(128) void k_flash_attn(
    const bf16* __restrict__ Q, const bf16* __restrict__ K,
    const bf16* __restrict__ V, bf16* __restrict__ O) {
  __shared__ __align__(16) bf16 Kst[32 * KROW];  // K block row-major (padded)
  __shared__ bf16 Vt[128 * 40];     // V block transposed: [d][key]
  __shared__ bf16 Pst[4][16 * 40];  // per-wave P staging (16 rows x 32 keys)

  const int tid  = threadIdx.x;
  const int h    = blockIdx.y;
  const int q0   = blockIdx.x * 64;
  const int kvh  = h >> 2;                      // GQA: 4 Q heads per KV head
  const bf16* Qh = Q + (size_t)h * HEADD;       // row stride NHEADS*HEADD
  const bf16* Kh = K + (size_t)kvh * HEADD;     // row stride NKVH*HEADD
  const bf16* Vh = V + (size_t)kvh * HEADD;
  bf16*       Oh = O + (size_t)h * HEADD;

  const int wave = tid >> 5, lane = tid & 31;
  const int half = lane >> 4, lrow = lane & 15;
  const int qb   = q0 + wave * 16;

#if USE_TDM
  const unsigned lds_kst = (unsigned)(uintptr_t)(void*)&Kst[0];
#endif

  // Q fragments (A layout), kept resident; scale already folded in by RoPE.
  v16bf qf[4];
#pragma unroll
  for (int kc = 0; kc < 4; ++kc) {
    const bf16* p = Qh + (size_t)(qb + lrow) * (NHEADS * HEADD)
                    + kc * 32 + half * 8;
    qf[kc] = load_frag2(p, p + 16);
  }

  v8f o[8] = {};
  float mi[8], li[8];
#pragma unroll
  for (int v = 0; v < 8; ++v) { mi[v] = -3.0e38f; li[v] = 0.f; }

  const int nkb  = (q0 >> 5) + 2;               // causal: keys <= q0+63
  const int vkey = tid >> 2, vds = tid & 3;     // V/K staging roles

  for (int kb = 0; kb < nkb; ++kb) {
    const int k0 = kb * 32;
    __syncthreads();  // prior iteration finished reading Kst / Vt / Pst

    // ---- stage K block [32 keys x 128 d] row-major into Kst
#if USE_TDM
    if (wave == 0) {
      // D# groups per CDNA5 ISA 8.3/8.4: 2D tensor, 2B elems, tile 128x32,
      // row pad 4 DWORDs every 64 DWORDs -> LDS row stride 272B.
      unsigned long long ga =
          (unsigned long long)(uintptr_t)(Kh + (size_t)k0 * (NKVH * HEADD));
      v4u g0 = { 1u,                                    // count=1, user mode
                 lds_kst,                               // lds_addr
                 (unsigned)ga,                          // global_addr[31:0]
                 (unsigned)((ga >> 32) & 0x1FFFFFFu) | (2u << 30) }; // +type=2
      v8i g1 = { (int)0x07510000,        // data_size=2B, pad_en, intv=64dw, amt=4dw
                 (int)(1024u << 16),     // tensor_dim0 = 1024 (row length)
                 (int)(2048u << 16),     // tensor_dim1 = 2048 (rows)
                 (int)(128u << 16),      // tile_dim0 = 128
                 32,                     // tile_dim1 = 32
                 1024,                   // tensor_dim0_stride = 1024 elems
                 0, 0 };                 // tensor_dim1_stride = 0 (2D)
      v4i z4 = { 0, 0, 0, 0 };
      v8i z8 = { 0, 0, 0, 0, 0, 0, 0, 0 };
      __builtin_amdgcn_tensor_load_to_lds(g0, g1, z4, z4, z8, 0);
    }
#else
    {
      const bf16* gk = Kh + (size_t)(k0 + vkey) * (NKVH * HEADD) + vds * 32;
      uint4* dst = reinterpret_cast<uint4*>(&Kst[vkey * KROW + vds * 32]);
      dst[0] = reinterpret_cast<const uint4*>(gk)[0];
      dst[1] = reinterpret_cast<const uint4*>(gk)[1];
    }
#endif

    // ---- stage V block transposed: Vt[d][key], 32 keys x 128 d
    {
      const bf16* gv = Vh + (size_t)(k0 + vkey) * (NKVH * HEADD) + vds * 32;
      H8 r[4];
#pragma unroll
      for (int j = 0; j < 4; ++j) r[j].u = reinterpret_cast<const uint4*>(gv)[j];
#pragma unroll
      for (int j = 0; j < 4; ++j)
#pragma unroll
        for (int i = 0; i < 8; ++i)
          Vt[(vds * 32 + j * 8 + i) * 40 + vkey] = r[j].h[i];
    }

#if USE_TDM
    if (wave == 0) __builtin_amdgcn_s_wait_tensorcnt(0);
#endif
    __syncthreads();  // Kst + Vt staged

    // ---- scores S = Q K^T (K rows are contraction-contiguous: B-frag layout)
    v8f sc[2] = {};
#pragma unroll
    for (int tj = 0; tj < 2; ++tj)
#pragma unroll
      for (int kc = 0; kc < 4; ++kc) {
        const bf16* p = &Kst[(tj * 16 + lrow) * KROW + kc * 32 + half * 8];
        v16bf kf = load_frag2(p, p + 16);
        sc[tj] = wmma_bf16(qf[kc], kf, sc[tj]);
      }

    // ---- causal mask (only near the diagonal)
    if (k0 + 31 > qb) {
#pragma unroll
      for (int tj = 0; tj < 2; ++tj)
#pragma unroll
        for (int v = 0; v < 8; ++v) {
          int qrow = qb + v + 8 * half;
          int key  = k0 + tj * 16 + lrow;
          if (key > qrow) sc[tj][v] = -1.0e30f;
        }
    }

    // ---- online softmax (row stats replicated across the 16-lane half)
#pragma unroll
    for (int v = 0; v < 8; ++v) {
      float bm = fmaxf(sc[0][v], sc[1][v]);
      bm = fmaxf(bm, __shfl_xor(bm, 1));
      bm = fmaxf(bm, __shfl_xor(bm, 2));
      bm = fmaxf(bm, __shfl_xor(bm, 4));
      bm = fmaxf(bm, __shfl_xor(bm, 8));
      float mnew  = fmaxf(mi[v], bm);
      float alpha = __expf(mi[v] - mnew);
      mi[v] = mnew;
      float p0 = __expf(sc[0][v] - mnew);
      float p1 = __expf(sc[1][v] - mnew);
      float rs = p0 + p1;
      rs += __shfl_xor(rs, 1);
      rs += __shfl_xor(rs, 2);
      rs += __shfl_xor(rs, 4);
      rs += __shfl_xor(rs, 8);
      li[v] = li[v] * alpha + rs;
#pragma unroll
      for (int j = 0; j < 8; ++j) o[j][v] *= alpha;
      int prow = v + 8 * half;                   // C-layout row
      Pst[wave][prow * 40 + lrow]      = (bf16)p0;
      Pst[wave][prow * 40 + 16 + lrow] = (bf16)p1;
    }
    __syncthreads();  // P visible (and all waves past Kst/Vt consumption point)

    // ---- O += P V  (P as A-frag via LDS, V as B-frags from Vt)
    {
      const bf16* pp = &Pst[wave][lrow * 40 + half * 8];
      v16bf pf = load_frag2(pp, pp + 16);
#pragma unroll
      for (int j = 0; j < 8; ++j) {
        const bf16* pv = &Vt[(j * 16 + lrow) * 40 + half * 8];
        v16bf vf = load_frag2(pv, pv + 16);
        o[j] = wmma_bf16(pf, vf, o[j]);
      }
    }
  }

  // ---- normalize and store (bf16, feeds final GEMM)
#pragma unroll
  for (int v = 0; v < 8; ++v) {
    float inv = 1.0f / li[v];
    int row = qb + v + 8 * half;
#pragma unroll
    for (int j = 0; j < 8; ++j)
      Oh[(size_t)row * (NHEADS * HEADD) + j * 16 + lrow] =
          (bf16)(o[j][v] * inv);
  }
}

// ---------------------------------------------------------------- host
extern "C" void kernel_launch(void* const* d_in, const int* in_sizes, int n_in,
                              void* d_out, int out_size, void* d_ws,
                              size_t ws_size, hipStream_t stream) {
  const float* hs   = (const float*)d_in[0];
  const float* wq   = (const float*)d_in[1];
  const float* wk   = (const float*)d_in[2];
  const float* wv   = (const float*)d_in[3];
  const float* wo   = (const float*)d_in[4];
  const float* cosb = (const float*)d_in[5];
  const float* sinb = (const float*)d_in[6];

  char* ws = (char*)d_ws;
  size_t off = 0;
  auto alloc = [&](size_t elems) -> bf16* {
    bf16* p = (bf16*)(ws + off);
    off = (off + elems * sizeof(bf16) + 255) & ~(size_t)255;
    return p;
  };

  bf16* X16  = alloc((size_t)S_LEN * HDIM);
  bf16* Wq16 = alloc((size_t)HDIM * (NHEADS * HEADD));
  bf16* Wk16 = alloc((size_t)HDIM * (NKVH * HEADD));
  bf16* Wv16 = alloc((size_t)HDIM * (NKVH * HEADD));
  bf16* Wo16 = alloc((size_t)(NHEADS * HEADD) * HDIM);
  bf16* Q16  = alloc((size_t)S_LEN * (NHEADS * HEADD));
  bf16* K16  = alloc((size_t)S_LEN * (NKVH * HEADD));
  bf16* V16  = alloc((size_t)S_LEN * (NKVH * HEADD));
  bf16* A16  = alloc((size_t)S_LEN * (NHEADS * HEADD));

  auto cvt = [&](const float* in, bf16* out, size_t n) {
    int n4 = (int)(n / 4);
    k_f32_to_bf16<<<(n4 + 255) / 256, 256, 0, stream>>>(in, out, n4);
  };
  cvt(hs, X16, (size_t)S_LEN * HDIM);
  cvt(wq, Wq16, (size_t)HDIM * (NHEADS * HEADD));
  cvt(wk, Wk16, (size_t)HDIM * (NKVH * HEADD));
  cvt(wv, Wv16, (size_t)HDIM * (NKVH * HEADD));
  cvt(wo, Wo16, (size_t)(NHEADS * HEADD) * HDIM);

  // Projections
  dim3 gq((NHEADS * HEADD) / 128, S_LEN / 128);
  k_gemm_bf16<false><<<gq, 256, 0, stream>>>(X16, Wq16, Q16,
                                             S_LEN, NHEADS * HEADD, HDIM);
  dim3 gkv((NKVH * HEADD) / 128, S_LEN / 128);
  k_gemm_bf16<false><<<gkv, 256, 0, stream>>>(X16, Wk16, K16,
                                              S_LEN, NKVH * HEADD, HDIM);
  k_gemm_bf16<false><<<gkv, 256, 0, stream>>>(X16, Wv16, V16,
                                              S_LEN, NKVH * HEADD, HDIM);

  // RoPE (fold 1/sqrt(HD) into Q)
  const float scale = 0.08838834764831845f;  // 1/sqrt(128)
  k_rope<<<(S_LEN * NHEADS * 64) / 256, 256, 0, stream>>>(Q16, cosb, sinb,
                                                          NHEADS, scale);
  k_rope<<<(S_LEN * NKVH * 64) / 256, 256, 0, stream>>>(K16, cosb, sinb,
                                                        NKVH, 1.0f);

  // Flash attention
  dim3 ga(S_LEN / 64, NHEADS);
  k_flash_attn<<<ga, 128, 0, stream>>>(Q16, K16, V16, A16);

  // Output projection -> f32 d_out
  dim3 go(HDIM / 128, S_LEN / 128);
  k_gemm_bf16<true><<<go, 256, 0, stream>>>(A16, Wo16, d_out,
                                            S_LEN, HDIM, HDIM);
}
